// GIN_MLP_31172872634624
// MI455X (gfx1250) — matrix operand order repaired and checked by reading, not compiled, not run on hardware
//
#include <hip/hip_runtime.h>
#include <hip/hip_bf16.h>
#include <cstdint>

// ---------------------------------------------------------------------------
// GIN link-prediction pipeline for gfx1250 (MI455X), wave32 + WMMA bf16.
//   conv1: agg -> GEMM(128,256)+BN+ReLU -> GEMM(256,128)+BN+ReLU
//   conv2: agg -> GEMM(128,256)+BN+ReLU -> GEMM(256,64)+BN
//   decoder: gather-concat -> GEMM(128,128)+bias+ReLU -> dot(128->1)
// A tiles are staged global->LDS with GLOBAL_LOAD_ASYNC_TO_LDS_B128
// (ASYNCcnt-tracked LDS-DMA) using double-buffered LDS so the DMA for tile
// c+1 overlaps the WMMA chain on tile c. bf16 conversion happens at fragment
// build (float2 LDS read -> v_cvt_pk_bf16_f32).
// ---------------------------------------------------------------------------

typedef __attribute__((ext_vector_type(16))) __bf16 v16bf;
typedef __attribute__((ext_vector_type(8)))  float  v8f;

union BPair { unsigned u; __bf16 h[2]; };

#define BN_EPS 1e-5f

// ---- fold BatchNorm (+ preceding linear bias) into per-channel affine ------
__global__ void make_affine_kernel(float* __restrict__ sc, float* __restrict__ sh,
                                   const float* __restrict__ g, const float* __restrict__ beta,
                                   const float* __restrict__ rm, const float* __restrict__ rv,
                                   const float* __restrict__ bias, int C) {
    int c = blockIdx.x * blockDim.x + threadIdx.x;
    if (c >= C) return;
    float b0 = bias ? bias[c] : 0.0f;
    if (g) {
        float s = g[c] * rsqrtf(rv[c] + BN_EPS);
        sc[c] = s;
        sh[c] = (b0 - rm[c]) * s + beta[c];
    } else {
        sc[c] = 1.0f;
        sh[c] = b0;
    }
}

// ---- agg init: h0 = (1+eps)*x ---------------------------------------------
__global__ void scaled_copy_kernel(const float* __restrict__ x, const float* __restrict__ epsp,
                                   float* __restrict__ y, size_t n) {
    size_t i = (size_t)blockIdx.x * blockDim.x + threadIdx.x;
    if (i >= n) return;
    y[i] = (1.0f + epsp[0]) * x[i];
}

// ---- edge scatter-add: agg[dst] += feat[src], F=128, one wave per edge ----
__global__ void scatter_add_kernel(const float* __restrict__ feat,
                                   const int* __restrict__ src, const int* __restrict__ dst,
                                   float* __restrict__ agg, int E) {
    int g    = blockIdx.x * blockDim.x + threadIdx.x;
    int e    = g >> 5;
    int lane = g & 31;
    if (e >= E) return;
    int s = src[e];
    int d = dst[e];
    const float4 v = *(const float4*)(feat + (size_t)s * 128 + lane * 4);
    float* p = agg + (size_t)d * 128 + lane * 4;
    atomicAdd(p + 0, v.x);
    atomicAdd(p + 1, v.y);
    atomicAdd(p + 2, v.z);
    atomicAdd(p + 3, v.w);
}

// ---- WMMA GEMM: C[M,Nc] = act( (A[M,K] @ W[K,Nc]) * sc + sh ) -------------
// Block = 256 threads = 8 waves; tile 128(M) x 32(N); each wave owns two
// 16x16 f32 accumulators via v_wmma_f32_16x16x32_bf16.
// GATHER: row r of A = concat(Z[gsrc[r]], Z[gdst[r]]), K/2 features each.
template <bool GATHER, bool RELU>
__global__ __launch_bounds__(256)
void gemm_bn_kernel(const float* __restrict__ A, const float* __restrict__ W,
                    const float* __restrict__ sc, const float* __restrict__ sh,
                    float* __restrict__ C, int M, int K, int Nc,
                    const int* __restrict__ gsrc, const int* __restrict__ gdst) {
    __shared__ float  As32[2][128][36];  // f32 A tiles; row stride 144B (16B-aligned)
    __shared__ __bf16 Bs[2][32][36];     // TRANSPOSED bf16: [n][k], pairs 32-bit

    const int tid  = threadIdx.x;
    const int lane = tid & 31;
    const int wave = tid >> 5;
    const int wm   = wave & 3;   // m sub-tile group: rows wm*32 .. wm*32+31
    const int wn   = wave >> 2;  // n sub-tile: cols wn*16 .. wn*16+15
    const int half = lane >> 4;
    const int hm   = lane & 15;

    const int rowBase = blockIdx.y * 128;
    const int colBase = blockIdx.x * 32;
    const int Kh      = K >> 1;  // gather mode: features per endpoint
    const int NCH     = K >> 5;  // number of 32-wide K chunks

    // B-tile load coords (one float4 per thread, coalesced along n)
    const int bkk = tid >> 3;        // 0..31: k within chunk
    const int bn4 = (tid & 7) * 4;   // 0,4,..,28: n within tile

    // ---- async DMA: stage A tile for K-chunk k0 into LDS buffer b ----
    auto issueA = [&](int k0, int b) {
        const int* idxp = nullptr;
        int        foff = 0;
        if (GATHER) {  // endpoint choice uniform per 32-wide chunk (K/2 % 32 == 0)
            const bool lo = (k0 < Kh);
            idxp = lo ? gsrc : gdst;
            foff = lo ? k0 : (k0 - Kh);
        }
#pragma unroll
        for (int i = 0; i < 4; ++i) {
            int seg = i * 256 + tid;
            int rr  = seg >> 3;           // 0..127
            int c4  = seg & 7;            // b128 column
            int gr  = min(rowBase + rr, M - 1);   // clamp; OOB rows masked at store
            const float* gp;
            if (GATHER) {
                gp = A + (size_t)idxp[gr] * Kh + foff + c4 * 4;
            } else {
                gp = A + (size_t)gr * K + k0 + c4 * 4;
            }
            unsigned lp = (unsigned)(uintptr_t)&As32[b][rr][c4 * 4];  // LDS byte offset
            asm volatile("global_load_async_to_lds_b128 %0, %1, off"
                         :: "v"(lp), "v"(gp) : "memory");
        }
    };
    // ---- stage B tile (f32 load -> bf16 transposed stores) ----
    auto stageB = [&](int k0, int b) {
        const float4 wv = *(const float4*)&W[(size_t)(k0 + bkk) * Nc + colBase + bn4];
        Bs[b][bn4 + 0][bkk] = (__bf16)wv.x;
        Bs[b][bn4 + 1][bkk] = (__bf16)wv.y;
        Bs[b][bn4 + 2][bkk] = (__bf16)wv.z;
        Bs[b][bn4 + 3][bkk] = (__bf16)wv.w;
    };

    v8f acc[2] = {};

    // ---- software pipeline: DMA(c+1) overlaps WMMA(c) ----
    issueA(0, 0);
    stageB(0, 0);
    for (int c = 0; c < NCH; ++c) {
        const int b = c & 1;
        // Only tile c's 4 async ops are outstanding here (c+1 not yet issued).
        asm volatile("s_wait_asynccnt 0x0" ::: "memory");
        __syncthreads();   // buffer b fully populated; all waves past reads of b^1

        if (c + 1 < NCH) {           // kick off next tile into the other buffer
            issueA((c + 1) * 32, b ^ 1);
            stageB((c + 1) * 32, b ^ 1);
        }

        // ---- B fragment (32x16: n = lane&15, K split by lane half) ----
        v16bf bf;
#pragma unroll
        for (int r = 0; r < 8; ++r) {
            BPair p;
            p.u = *(const unsigned*)&Bs[b][wn * 16 + hm][half * 16 + 2 * r];
            bf[2 * r]     = p.h[0];
            bf[2 * r + 1] = p.h[1];
        }
        // ---- two A fragments (16x32) from f32 LDS, cvt_pk at build ----
#pragma unroll
        for (int t = 0; t < 2; ++t) {
            const int m0 = wm * 32 + t * 16;
            v16bf af;
#pragma unroll
            for (int r = 0; r < 8; ++r) {
                int kb = half * 16 + r * 2;
                float2 f = *(const float2*)&As32[b][m0 + hm][kb];
                af[2 * r]     = (__bf16)f.x;
                af[2 * r + 1] = (__bf16)f.y;
            }
            acc[t] = __builtin_amdgcn_wmma_f32_16x16x32_bf16(
                false, af, false, bf, (short)0, acc[t], false, false);
        }
        // no trailing barrier: next iteration writes only the other buffer,
        // and the next top-of-loop barrier fences before buffer reuse.
    }

    // ---- epilogue: per-channel affine (+ReLU) ----
    const int col = colBase + wn * 16 + hm;  // C/D layout: N = lane&15
    const float s = sc[col];
    const float h = sh[col];
#pragma unroll
    for (int t = 0; t < 2; ++t) {
        const int r0 = rowBase + wm * 32 + t * 16 + 8 * half;  // M = r + 8*half
        float* p = C + (size_t)r0 * Nc + col;
        if (r0 + 8 <= M) {
#pragma unroll
            for (int r = 0; r < 8; ++r) {
                float y = acc[t][r] * s + h;
                if (RELU) y = fmaxf(y, 0.0f);
                p[(size_t)r * Nc] = y;
            }
        } else {
#pragma unroll
            for (int r = 0; r < 8; ++r) {
                if (r0 + r < M) {
                    float y = acc[t][r] * s + h;
                    if (RELU) y = fmaxf(y, 0.0f);
                    p[(size_t)r * Nc] = y;
                }
            }
        }
    }
}

// ---- final projection: out[e] = dot(T[e,:128], w) + b; one wave per row ---
__global__ void dot_out_kernel(const float* __restrict__ T, const float* __restrict__ w,
                               const float* __restrict__ b, float* __restrict__ out, int M) {
    int g    = blockIdx.x * blockDim.x + threadIdx.x;
    int row  = g >> 5;
    int lane = g & 31;
    if (row >= M) return;
    const float4 a  = *(const float4*)(T + (size_t)row * 128 + lane * 4);
    const float4 wv = *(const float4*)(w + lane * 4);
    float s = a.x * wv.x + a.y * wv.y + a.z * wv.z + a.w * wv.w;
#pragma unroll
    for (int off = 16; off > 0; off >>= 1)
        s += __shfl_xor(s, off, 32);
    if (lane == 0) out[row] = s + b[0];
}

// ---------------------------------------------------------------------------
extern "C" void kernel_launch(void* const* d_in, const int* in_sizes, int n_in,
                              void* d_out, int out_size, void* d_ws, size_t ws_size,
                              hipStream_t stream) {
    // ---- inputs (setup_inputs order) ----
    const float* x    = (const float*)d_in[0];
    const int*   ei   = (const int*)  d_in[1];
    const int*   eli  = (const int*)  d_in[2];
    const float* eps1 = (const float*)d_in[3];
    const float* W1a  = (const float*)d_in[4];
    const float* b1a  = (const float*)d_in[5];
    const float* g1a  = (const float*)d_in[6];
    const float* be1a = (const float*)d_in[7];
    const float* rm1a = (const float*)d_in[8];
    const float* rv1a = (const float*)d_in[9];
    const float* W1b  = (const float*)d_in[10];
    const float* b1b  = (const float*)d_in[11];
    const float* g1   = (const float*)d_in[12];
    const float* be1  = (const float*)d_in[13];
    const float* rm1  = (const float*)d_in[14];
    const float* rv1  = (const float*)d_in[15];
    const float* eps2 = (const float*)d_in[16];
    const float* W2a  = (const float*)d_in[17];
    const float* b2a  = (const float*)d_in[18];
    const float* g2a  = (const float*)d_in[19];
    const float* be2a = (const float*)d_in[20];
    const float* rm2a = (const float*)d_in[21];
    const float* rv2a = (const float*)d_in[22];
    const float* W2b  = (const float*)d_in[23];
    const float* b2b  = (const float*)d_in[24];
    const float* g2   = (const float*)d_in[25];
    const float* be2  = (const float*)d_in[26];
    const float* rm2  = (const float*)d_in[27];
    const float* rv2  = (const float*)d_in[28];
    const float* Wd1  = (const float*)d_in[29];
    const float* bd1  = (const float*)d_in[30];
    const float* Wd2  = (const float*)d_in[31];
    const float* bd2  = (const float*)d_in[32];

    const int N  = in_sizes[0] / 128;
    const int E  = in_sizes[1] / 2;
    const int EL = in_sizes[2] / 2;
    const int H = 128, Dout = 64;

    const int* e_src  = ei;            // edge_index[0]
    const int* e_dst  = ei + E;        // edge_index[1]
    const int* l_src  = eli;           // edge_label_index[0]
    const int* l_dst  = eli + EL;      // edge_label_index[1]

    // ---- workspace carve (floats) ----
    float* ws = (float*)d_ws;
    size_t o = 0;
    float* sc1a = ws + o; o += 256;  float* sh1a = ws + o; o += 256;
    float* sc1b = ws + o; o += 128;  float* sh1b = ws + o; o += 128;
    float* sc2a = ws + o; o += 256;  float* sh2a = ws + o; o += 256;
    float* sc2b = ws + o; o += 64;   float* sh2b = ws + o; o += 64;
    float* scd1 = ws + o; o += 128;  float* shd1 = ws + o; o += 128;
    o = (o + 63) & ~(size_t)63;
    float* h0  = ws + o; o += (size_t)N * 128;   // agg1 / h0, reused as agg2
    float* t1  = ws + o; o += (size_t)N * 256;   // hidden 2H, reused for conv2
    float* h1  = ws + o; o += (size_t)N * 128;   // conv1 output (post bn1+relu)
    float* z   = ws + o; o += (size_t)N * 64;    // conv2 output (post bn2)
    float* td1 = ws + o; o += (size_t)EL * 128;  // decoder hidden

    float* out = (float*)d_out;

    const dim3 blk(256);

    // ---- fold BN / bias into per-channel affine ----
    make_affine_kernel<<<1, 256, 0, stream>>>(sc1a, sh1a, g1a, be1a, rm1a, rv1a, b1a, 2 * H);
    make_affine_kernel<<<1, 128, 0, stream>>>(sc1b, sh1b, g1,  be1,  rm1,  rv1,  b1b, H);
    make_affine_kernel<<<1, 256, 0, stream>>>(sc2a, sh2a, g2a, be2a, rm2a, rv2a, b2a, 2 * H);
    make_affine_kernel<<<1, 64,  0, stream>>>(sc2b, sh2b, g2,  be2,  rm2,  rv2,  b2b, Dout);
    make_affine_kernel<<<1, 128, 0, stream>>>(scd1, shd1, nullptr, nullptr, nullptr, nullptr, bd1, H);

    // ---- conv1 aggregation: h0 = (1+eps1)*x + sum_{j->i} x[j] ----
    {
        size_t n = (size_t)N * 128;
        scaled_copy_kernel<<<(unsigned)((n + 255) / 256), blk, 0, stream>>>(x, eps1, h0, n);
        size_t th = (size_t)E * 32;
        scatter_add_kernel<<<(unsigned)((th + 255) / 256), blk, 0, stream>>>(x, e_src, e_dst, h0, E);
    }
    // ---- conv1 MLP ----
    {
        dim3 grid(256 / 32, (N + 127) / 128);
        gemm_bn_kernel<false, true><<<grid, blk, 0, stream>>>(h0, W1a, sc1a, sh1a, t1,
                                                              N, 128, 256, nullptr, nullptr);
        dim3 grid2(128 / 32, (N + 127) / 128);
        gemm_bn_kernel<false, true><<<grid2, blk, 0, stream>>>(t1, W1b, sc1b, sh1b, h1,
                                                               N, 256, 128, nullptr, nullptr);
    }
    // ---- conv2 aggregation (reuse h0 as agg2) ----
    {
        size_t n = (size_t)N * 128;
        scaled_copy_kernel<<<(unsigned)((n + 255) / 256), blk, 0, stream>>>(h1, eps2, h0, n);
        size_t th = (size_t)E * 32;
        scatter_add_kernel<<<(unsigned)((th + 255) / 256), blk, 0, stream>>>(h1, e_src, e_dst, h0, E);
    }
    // ---- conv2 MLP ----
    {
        dim3 grid(256 / 32, (N + 127) / 128);
        gemm_bn_kernel<false, true><<<grid, blk, 0, stream>>>(h0, W2a, sc2a, sh2a, t1,
                                                              N, 128, 256, nullptr, nullptr);
        dim3 grid2(64 / 32, (N + 127) / 128);
        gemm_bn_kernel<false, false><<<grid2, blk, 0, stream>>>(t1, W2b, sc2b, sh2b, z,
                                                                N, 256, 64, nullptr, nullptr);
    }
    // ---- decoder: gather-concat GEMM + ReLU, then dot(128->1) ----
    {
        dim3 grid(128 / 32, (EL + 127) / 128);
        gemm_bn_kernel<true, true><<<grid, blk, 0, stream>>>(z, Wd1, scd1, shd1, td1,
                                                             EL, 128, 128, l_src, l_dst);
        size_t th = (size_t)EL * 32;
        dot_out_kernel<<<(unsigned)((th + 255) / 256), blk, 0, stream>>>(td1, Wd2, bd2, out, EL);
    }
}
